// GFusion_19533511262231
// MI455X (gfx1250) — matrix operand order repaired and checked by reading, not compiled, not run on hardware
//
#include <hip/hip_runtime.h>

typedef __attribute__((ext_vector_type(16))) _Float16 v16h;
typedef __attribute__((ext_vector_type(8)))  _Float16 v8h;
typedef __attribute__((ext_vector_type(4)))  _Float16 v4h;
typedef __attribute__((ext_vector_type(8)))  float    v8f;

#define WMMA(a, b, c) \
  __builtin_amdgcn_wmma_f32_16x16x32_f16(false, (a), false, (b), (short)0, (c), false, false)

namespace {
constexpr int   D      = 384;
constexpr int   NQ     = 8192;
constexpr int   KT     = 32;                       // keys per flash tile
constexpr int   NT     = NQ / KT;                  // 256 key tiles
constexpr int   LDR    = 392;                      // padded leading dim (halfs) for 384-wide LDS tiles
constexpr int   LDT    = 40;                       // padded leading dim for V-transposed / P tiles
constexpr float SCALE2 = 0.07362229f;              // (1/sqrt(384)) * log2(e)
}  // namespace

// CDNA5 async global->LDS copy, 16 bytes per lane, tracked by ASYNCcnt.
// A generic pointer to LDS carries the LDS byte offset in its low 32 bits (ISA 10.2),
// which is exactly the VDST operand of the async op.
__device__ __forceinline__ void async_b128(const void* gp, void* lp) {
  asm volatile("global_load_async_to_lds_b128 %0, %1, off"
               :
               : "v"((unsigned)(uintptr_t)lp), "v"((unsigned long long)(uintptr_t)gp)
               : "memory");
}
__device__ __forceinline__ void wait_async_le12() {
  asm volatile("s_wait_asynccnt 0xc" ::: "memory");
}
__device__ __forceinline__ void wait_async_0() {
  asm volatile("s_wait_asynccnt 0x0" ::: "memory");
}

// A-matrix (16x32 f16) per-lane load in this file's contiguous k order (not the ISA table order):
// lane L: row m = L&15; VGPRs 0..7 hold K = (L>=16?16:0)+0..15 (contiguous, matching B operands)
__device__ __forceinline__ v16h load_a16(const _Float16* tile, int ld, int kbase, int lane) {
  const int m  = lane & 15;
  const int h8 = (lane >> 4) << 4;
  const _Float16* p = tile + m * ld + kbase + h8;
  union { v16h v; v8h h[2]; } u;
  u.h[0] = *reinterpret_cast<const v8h*>(p);
  u.h[1] = *reinterpret_cast<const v8h*>(p + 8);
  return u.v;
}

__global__ void k_cvt_f16(const float* __restrict__ src, _Float16* __restrict__ dst, int n4) {
  const int i = blockIdx.x * blockDim.x + threadIdx.x;
  if (i >= n4) return;
  const float4 f = reinterpret_cast<const float4*>(src)[i];
  v4h h;
  h[0] = (_Float16)f.x; h[1] = (_Float16)f.y; h[2] = (_Float16)f.z; h[3] = (_Float16)f.w;
  reinterpret_cast<v4h*>(dst)[i] = h;
}

// Out[r][c] = In[r][:] . W[c][:] + bias[c]   (W row-major [out][in], i.e. X @ W^T + b)
template <bool OUT_F16>
__global__ __launch_bounds__(128) void k_proj(const _Float16* __restrict__ In,
                                              const _Float16* __restrict__ W,
                                              const float* __restrict__ bias,
                                              void* __restrict__ Out) {
  __shared__ _Float16 InS[16 * LDR];
  const int tid  = threadIdx.x;
  const int lane = tid & 31;
  const int wave = tid >> 5;
  const int rbase = blockIdx.x << 4;

  for (int i = tid; i < 16 * 48; i += 128) {
    const int r = i / 48, ch = i % 48;
    async_b128(&In[(size_t)(rbase + r) * D + ch * 8], &InS[r * LDR + ch * 8]);
  }
  wait_async_0();
  __syncthreads();

  const int khalf = (lane >> 4) << 4;
  const int nidx  = lane & 15;
#pragma unroll
  for (int i = 0; i < 6; ++i) {
    const int cbase = (wave * 6 + i) << 4;
    v8f c = {};
#pragma unroll
    for (int kk = 0; kk < 12; ++kk) {
      const v16h a = load_a16(InS, LDR, kk * 32, lane);
      // B[k][n] = W[cbase+n][kk*32+k]: lane reads 16 contiguous halfs of one weight row
      const v16h b = *reinterpret_cast<const v16h*>(&W[(size_t)(cbase + nidx) * D + kk * 32 + khalf]);
      c = WMMA(a, b, c);
    }
    const float bv   = bias[cbase + nidx];
    const int   colg = cbase + nidx;
#pragma unroll
    for (int v = 0; v < 8; ++v) {
      const int   rowg = rbase + ((lane >> 4) << 3) + v;
      const float val  = c[v] + bv;
      if (OUT_F16)
        reinterpret_cast<_Float16*>(Out)[(size_t)rowg * D + colg] = (_Float16)val;
      else
        reinterpret_cast<float*>(Out)[(size_t)rowg * D + colg] = val;
    }
  }
}

// Flash attention + 1/l + row L2-normalize + residual, emits Z = x + normalize(ctx) as f16.
// Block = 128 threads (4 waves); each wave owns 16 query rows; block shares K/V staging.
// K tiles are double-buffered via async global->LDS DMA overlapping the WMMA work.
__global__ __launch_bounds__(128) void k_flash(const _Float16* __restrict__ Qh,
                                               const _Float16* __restrict__ Kh,
                                               const _Float16* __restrict__ Vh,
                                               const float* __restrict__ Xres,
                                               _Float16* __restrict__ Zh) {
  __shared__ _Float16 Qs[4 * 16 * LDR];   // 64 query rows, row-major
  __shared__ _Float16 Ks[2][KT * LDR];    // ping-pong 32-key tiles, row-major
  __shared__ _Float16 Vs[D * LDT];        // V transposed: [d][key]
  __shared__ _Float16 Ps[4 * 16 * LDT];   // per-wave P bounce (C-layout -> A-layout)

  const int tid   = threadIdx.x;
  const int lane  = tid & 31;
  const int wave  = tid >> 5;
  const int nidx  = lane & 15;
  const int hi    = lane >> 4;
  const int khalf = hi << 4;
  const int qblk  = blockIdx.x * 64;

  // async-stage Q (64 rows): completes before the first in-loop wait/barrier
  for (int i = tid; i < 64 * 48; i += 128) {
    const int r = i / 48, ch = i % 48;
    async_b128(&Qh[(size_t)(qblk + r) * D + ch * 8], &Qs[r * LDR + ch * 8]);
  }
  // async-stage K tile 0 into buffer 0
  for (int i = tid; i < KT * 48; i += 128) {
    const int r = i / 48, ch = i % 48;
    async_b128(&Kh[(size_t)r * D + ch * 8], &Ks[0][r * LDR + ch * 8]);
  }

  _Float16* Qw = &Qs[wave * 16 * LDR];
  _Float16* Pw = &Ps[wave * 16 * LDT];

  v8f acc[24];
#pragma unroll
  for (int j = 0; j < 24; ++j) acc[j] = {};
  float m_v[8], l_v[8];
#pragma unroll
  for (int v = 0; v < 8; ++v) { m_v[v] = -1e30f; l_v[v] = 0.0f; }

  for (int t = 0; t < NT; ++t) {
    const int buf = t & 1;
    // stage V tile t: register bounce + transpose scatter into Vs (protected by prev end barrier)
    const _Float16* Vg = Vh + (size_t)t * KT * D;
    for (int i = tid; i < KT * 48; i += 128) {
      const int r = i / 48, ch = i % 48;
      uint4 raw = *reinterpret_cast<const uint4*>(&Vg[r * D + ch * 8]);
      union { uint4 q; _Float16 e[8]; } u;
      u.q = raw;
      const int d0 = ch * 8;
#pragma unroll
      for (int e = 0; e < 8; ++e) Vs[(d0 + e) * LDT + r] = u.e[e];
    }
    // issue async K for the next tile into the other buffer (wraps to tile 0 at the end:
    // harmless, keeps the wait immediate constant; s_endpgm waits idle)
    {
      const _Float16* Kg = Kh + (size_t)((t + 1) & (NT - 1)) * KT * D;
      _Float16* dstb = Ks[buf ^ 1];
      for (int i = tid; i < KT * 48; i += 128) {
        const int r = i / 48, ch = i % 48;
        async_b128(&Kg[(size_t)r * D + ch * 8], &dstb[r * LDR + ch * 8]);
      }
    }
    if (t + 1 < NT) {  // warm caches for the next V tile while computing on this one
      for (int off = tid * 64; off < KT * D; off += 128 * 64)
        __builtin_prefetch(Vg + KT * D + off, 0, 1);
    }
    // async loads complete in order: <=12 outstanding ==> tile t's K (and Q) resident
    wait_async_le12();
    __syncthreads();

    // S = Q . K^T : two 16x16 C tiles covering 32 keys, K-dim = 384 in 12 steps
    const _Float16* Kb = Ks[buf];
    v8f c0 = {}, c1 = {};
#pragma unroll
    for (int kk = 0; kk < 12; ++kk) {
      const v16h a  = load_a16(Qw, LDR, kk * 32, lane);
      const v16h b0 = *reinterpret_cast<const v16h*>(&Kb[nidx * LDR + kk * 32 + khalf]);
      c0 = WMMA(a, b0, c0);
      const v16h b1 = *reinterpret_cast<const v16h*>(&Kb[(16 + nidx) * LDR + kk * 32 + khalf]);
      c1 = WMMA(a, b1, c1);
    }

    // Online softmax in exp2 domain. Row r = hi*8+v lives in one 16-lane half -> 16-lane butterflies.
    float alpha[8];
#pragma unroll
    for (int v = 0; v < 8; ++v) {
      const float s0 = c0[v] * SCALE2;
      const float s1 = c1[v] * SCALE2;
      float mx = fmaxf(s0, s1);
#pragma unroll
      for (int msk = 1; msk < 16; msk <<= 1) mx = fmaxf(mx, __shfl_xor(mx, msk, 32));
      const float mnew = fmaxf(m_v[v], mx);
      alpha[v] = exp2f(m_v[v] - mnew);
      m_v[v]   = mnew;
      const float p0 = exp2f(s0 - mnew);
      const float p1 = exp2f(s1 - mnew);
      float rs = p0 + p1;
#pragma unroll
      for (int msk = 1; msk < 16; msk <<= 1) rs += __shfl_xor(rs, msk, 32);
      l_v[v] = l_v[v] * alpha[v] + rs;
      const int r = (hi << 3) + v;
      Pw[r * LDT + nidx]      = (_Float16)p0;   // C-layout -> A-layout via LDS bounce
      Pw[r * LDT + 16 + nidx] = (_Float16)p1;
    }

    // ctx = alpha*ctx + P . V   (B[k][n] = V[key k][j*16+n], contiguous in transposed Vs)
    const v16h ap = load_a16(Pw, LDT, 0, lane);
#pragma unroll
    for (int j = 0; j < 24; ++j) {
      const v16h bv = *reinterpret_cast<const v16h*>(&Vs[(j * 16 + nidx) * LDT + khalf]);
      v8f a = acc[j];
#pragma unroll
      for (int v = 0; v < 8; ++v) a[v] *= alpha[v];
      acc[j] = WMMA(ap, bv, a);
    }
    __syncthreads();  // everyone done reading Ks[buf]/Vs before next iteration restages
  }

  // Epilogue: ctx /= l ; row L2-normalize ; add residual x ; store f16
  float rl[8], rn[8];
#pragma unroll
  for (int v = 0; v < 8; ++v) rl[v] = 1.0f / l_v[v];
#pragma unroll
  for (int v = 0; v < 8; ++v) {
    float s = 0.0f;
#pragma unroll
    for (int j = 0; j < 24; ++j) {
      const float val = acc[j][v] * rl[v];
      s += val * val;
    }
#pragma unroll
    for (int msk = 1; msk < 16; msk <<= 1) s += __shfl_xor(s, msk, 32);
    const float nrm = sqrtf(s);
    rn[v] = 1.0f / fmaxf(nrm, 1e-12f);
  }
  const int qrow0 = qblk + wave * 16 + (hi << 3);
#pragma unroll
  for (int j = 0; j < 24; ++j) {
    const int col = j * 16 + nidx;
#pragma unroll
    for (int v = 0; v < 8; ++v) {
      const int   row = qrow0 + v;
      const float val = acc[j][v] * rl[v] * rn[v] + Xres[(size_t)row * D + col];
      Zh[(size_t)row * D + col] = (_Float16)val;
    }
  }
}

extern "C" void kernel_launch(void* const* d_in, const int* in_sizes, int n_in,
                              void* d_out, int out_size, void* d_ws, size_t ws_size,
                              hipStream_t stream) {
  (void)in_sizes; (void)n_in; (void)out_size; (void)ws_size;
  const float* x  = (const float*)d_in[0];
  const float* y  = (const float*)d_in[1];
  const float* Wq = (const float*)d_in[2];
  const float* bq = (const float*)d_in[3];
  const float* Wk = (const float*)d_in[4];
  const float* bk = (const float*)d_in[5];
  const float* Wv = (const float*)d_in[6];
  const float* bv = (const float*)d_in[7];
  const float* Wo = (const float*)d_in[8];
  const float* bo = (const float*)d_in[9];

  char* p = (char*)d_ws;
  auto take = [&](size_t elems) -> _Float16* {
    _Float16* r = (_Float16*)p;
    p += (elems * sizeof(_Float16) + 255) & ~(size_t)255;
    return r;
  };
  const size_t ND = (size_t)NQ * D, WD = (size_t)D * D;
  _Float16* xh  = take(ND);
  _Float16* yh  = take(ND);
  _Float16* Qh  = take(ND);
  _Float16* Kh  = take(ND);
  _Float16* Vh  = take(ND);
  _Float16* Zh  = take(ND);
  _Float16* Wqh = take(WD);
  _Float16* Wkh = take(WD);
  _Float16* Wvh = take(WD);
  _Float16* Woh = take(WD);

  const int n4d = (int)(ND / 4), n4w = (int)(WD / 4);
  k_cvt_f16<<<(n4d + 255) / 256, 256, 0, stream>>>(x, xh, n4d);
  k_cvt_f16<<<(n4d + 255) / 256, 256, 0, stream>>>(y, yh, n4d);
  k_cvt_f16<<<(n4w + 255) / 256, 256, 0, stream>>>(Wq, Wqh, n4w);
  k_cvt_f16<<<(n4w + 255) / 256, 256, 0, stream>>>(Wk, Wkh, n4w);
  k_cvt_f16<<<(n4w + 255) / 256, 256, 0, stream>>>(Wv, Wvh, n4w);
  k_cvt_f16<<<(n4w + 255) / 256, 256, 0, stream>>>(Wo, Woh, n4w);

  k_proj<true><<<NQ / 16, 128, 0, stream>>>(xh, Wqh, bq, Qh);
  k_proj<true><<<NQ / 16, 128, 0, stream>>>(yh, Wkh, bk, Kh);
  k_proj<true><<<NQ / 16, 128, 0, stream>>>(yh, Wvh, bv, Vh);

  k_flash<<<NQ / 64, 128, 0, stream>>>(Qh, Kh, Vh, x, Zh);

  k_proj<false><<<NQ / 16, 128, 0, stream>>>(Zh, Woh, bo, d_out);
}
